// SelfAttention_61521111547908
// MI455X (gfx1250) — matrix operand order repaired and checked by reading, not compiled, not run on hardware
//
#include <hip/hip_runtime.h>

#define E_DIM 1024
#define HEADS 16
#define HDIM  64
#define SEQ   2048
#define BATCH 4
#define SCALE_Q 0.125f   // 64^-0.5

typedef __attribute__((ext_vector_type(16))) __bf16 v16bf;
typedef __attribute__((ext_vector_type(8)))  float  v8f;

union ABu { v16bf v; uint4 q[2]; };

// 16x f32 -> v16bf via native v_cvt_pk_bf16_f32
__device__ __forceinline__ v16bf cvt16(float4 a, float4 b, float4 c, float4 d) {
  v16bf r;
  r[0]  = (__bf16)a.x; r[1]  = (__bf16)a.y; r[2]  = (__bf16)a.z; r[3]  = (__bf16)a.w;
  r[4]  = (__bf16)b.x; r[5]  = (__bf16)b.y; r[6]  = (__bf16)b.z; r[7]  = (__bf16)b.w;
  r[8]  = (__bf16)c.x; r[9]  = (__bf16)c.y; r[10] = (__bf16)c.z; r[11] = (__bf16)c.w;
  r[12] = (__bf16)d.x; r[13] = (__bf16)d.y; r[14] = (__bf16)d.z; r[15] = (__bf16)d.w;
  return r;
}

// ---- WMMA tile loaders -------------------------------------------------
// A (16x32 bf16): lane holds row m=lane&15; element j -> K = (j/8)*16 + (lane>>4)*8 + j%8
__device__ __forceinline__ v16bf load_a_f32(const float* __restrict__ src, int ld, int lane) {
  int m = lane & 15, kb = (lane >> 4) * 8;
  const float* p0 = src + (size_t)m * ld + kb;
  const float* p1 = p0 + 16;
  float4 a = *(const float4*)p0, b = *(const float4*)(p0 + 4);
  float4 c = *(const float4*)p1, d = *(const float4*)(p1 + 4);
  return cvt16(a, b, c, d);
}
__device__ __forceinline__ v16bf load_a_bf16(const __bf16* __restrict__ src, int ld, int lane) {
  int m = lane & 15, kb = (lane >> 4) * 8;
  const __bf16* p0 = src + (size_t)m * ld + kb;
  ABu u;
  u.q[0] = *(const uint4*)p0;
  u.q[1] = *(const uint4*)(p0 + 16);
  return u.v;
}
// B (32x16 bf16): lane holds col n=lane&15; element j -> K = (j/8)*16 + (lane>>4)*8 + j%8
// source is row-major R[n][k]  (i.e. B[k][n] = R[n][k]): two 16-byte loads per lane, 32 bytes apart.
__device__ __forceinline__ v16bf load_b_bf16(const __bf16* __restrict__ src, int ld, int lane) {
  int n = lane & 15, k0 = (lane >> 4) * 8;
  const __bf16* p = src + (size_t)n * ld + k0;
  ABu u;
  u.q[0] = *(const uint4*)p;
  u.q[1] = *(const uint4*)(p + 16);
  return u.v;
}

__device__ __forceinline__ v8f wmma_bf16(v16bf a, v16bf b, v8f c) {
  return __builtin_amdgcn_wmma_f32_16x16x32_bf16(false, a, false, b, (short)0, c, false, false);
}

// ---- Kernel 0a: comb = mask + bias -------------------------------------
__global__ void combine_kernel(const float* __restrict__ bias,
                               const float* __restrict__ mask,
                               float* __restrict__ comb, int n4) {
  int i = blockIdx.x * blockDim.x + threadIdx.x;
  if (i < n4) {
    float4 a = ((const float4*)bias)[i];
    float4 b = ((const float4*)mask)[i];
    float4 o;
    o.x = a.x + b.x; o.y = a.y + b.y; o.z = a.z + b.z; o.w = a.w + b.w;
    ((float4*)comb)[i] = o;
  }
}

// ---- Kernel 0b: weight f32 -> bf16 pre-conversion ----------------------
__global__ void w_to_bf16_kernel(const float* __restrict__ src,
                                 __bf16* __restrict__ dst, int n4) {
  int i = blockIdx.x * blockDim.x + threadIdx.x;
  if (i < n4) {
    float4 v = ((const float4*)src)[i];
    union { __bf16 h[4]; uint2 u; } p;
    p.h[0] = (__bf16)v.x; p.h[1] = (__bf16)v.y;
    p.h[2] = (__bf16)v.z; p.h[3] = (__bf16)v.w;
    ((uint2*)dst)[i] = p.u;
  }
}

// ---- Kernel 1: fused QKV projection (WMMA bf16, ping-pong pipelined) ---
// grid (128, 16, 3), block 128. Wave computes 16(m) x 64(n); n-block == one head.
__global__ __launch_bounds__(128) void qkv_proj_kernel(
    const float* __restrict__ query, const float* __restrict__ key,
    const float* __restrict__ value, const __bf16* __restrict__ Wb,
    const float* __restrict__ bqkv,
    __bf16* __restrict__ qw, __bf16* __restrict__ kw,
    __bf16* __restrict__ vtw) {
  int lane = threadIdx.x & 31, wave = threadIdx.x >> 5;
  int m0 = (blockIdx.x * 4 + wave) * 16;   // row in [0, 8192)
  int n0 = blockIdx.y * 64;
  int proj = blockIdx.z;
  const float*  X  = ((proj == 0) ? query : (proj == 1 ? key : value)) + (size_t)m0 * E_DIM;
  const __bf16* W0 = Wb + (size_t)proj * E_DIM * E_DIM + (size_t)n0 * E_DIM;

  v8f acc0 = {}, acc1 = {}, acc2 = {}, acc3 = {};

#define LOAD_SET(AV, B0, B1, B2, B3, KK)                                   \
  AV = load_a_f32 (X + (KK), E_DIM, lane);                                 \
  B0 = load_b_bf16(W0 + (size_t) 0 * E_DIM + (KK), E_DIM, lane);           \
  B1 = load_b_bf16(W0 + (size_t)16 * E_DIM + (KK), E_DIM, lane);           \
  B2 = load_b_bf16(W0 + (size_t)32 * E_DIM + (KK), E_DIM, lane);           \
  B3 = load_b_bf16(W0 + (size_t)48 * E_DIM + (KK), E_DIM, lane);
#define DO_STEP(AV, B0, B1, B2, B3)                                        \
  acc0 = wmma_bf16(AV, B0, acc0);                                          \
  acc1 = wmma_bf16(AV, B1, acc1);                                          \
  acc2 = wmma_bf16(AV, B2, acc2);                                          \
  acc3 = wmma_bf16(AV, B3, acc3);

  v16bf aP, bP0, bP1, bP2, bP3;   // ping
  v16bf aQ, bQ0, bQ1, bQ2, bQ3;   // pong
  LOAD_SET(aP, bP0, bP1, bP2, bP3, 0)
  int kk = 32;
  for (int it = 0; it < 15; ++it, kk += 64) {
    LOAD_SET(aQ, bQ0, bQ1, bQ2, bQ3, kk)
    DO_STEP(aP, bP0, bP1, bP2, bP3)            // step kk-32
    LOAD_SET(aP, bP0, bP1, bP2, bP3, kk + 32)
    DO_STEP(aQ, bQ0, bQ1, bQ2, bQ3)            // step kk
  }
  LOAD_SET(aQ, bQ0, bQ1, bQ2, bQ3, 992)
  DO_STEP(aP, bP0, bP1, bP2, bP3)              // step 960
  DO_STEP(aQ, bQ0, bQ1, bQ2, bQ3)              // step 992
#undef LOAD_SET
#undef DO_STEP

  int nloc = lane & 15, hi = lane >> 4;
  int b = m0 >> 11, l0 = m0 & 2047, h = n0 >> 6;
  float scale = (proj == 0) ? SCALE_Q : 1.0f;
  size_t bh = (size_t)(b * HEADS + h);
#define QKV_STORE(T, ACC)                                                       \
  {                                                                             \
    float bias = bqkv[(size_t)proj * E_DIM + n0 + (T) * 16 + nloc];             \
    int d = (T) * 16 + nloc;                                                    \
    for (int r = 0; r < 8; ++r) {                                               \
      float v = (ACC[r] + bias) * scale;                                        \
      int row = l0 + r + hi * 8;                                                \
      if (proj == 0)      qw [(bh * SEQ + row) * HDIM + d] = (__bf16)v;         \
      else if (proj == 1) kw [(bh * SEQ + row) * HDIM + d] = (__bf16)v;         \
      else                vtw[(bh * HDIM + d) * SEQ + row] = (__bf16)v;         \
    }                                                                           \
  }
  QKV_STORE(0, acc0) QKV_STORE(1, acc1) QKV_STORE(2, acc2) QKV_STORE(3, acc3)
#undef QKV_STORE
}

// ---- Kernel 2: flash attention (online softmax, WMMA bf16, pipelined) --
// grid (32, 64), block 128. Wave handles 16 query rows of one (b,h).
__global__ __launch_bounds__(128) void attn_kernel(
    const __bf16* __restrict__ qw, const __bf16* __restrict__ kw,
    const __bf16* __restrict__ vtw, const float* __restrict__ comb,
    __bf16* __restrict__ aw) {
  __shared__ __align__(16) __bf16 plds[4][16][40];  // padded rows: 80B stride
  int lane = threadIdx.x & 31, wave = threadIdx.x >> 5;
  int bh = blockIdx.y;
  int b = bh >> 4, h = bh & 15;
  int l0 = blockIdx.x * 64 + wave * 16;
  const __bf16* Q  = qw  + (size_t)bh * SEQ * HDIM;
  const __bf16* K  = kw  + (size_t)bh * SEQ * HDIM;
  const __bf16* Vt = vtw + (size_t)bh * HDIM * SEQ;
  int nloc = lane & 15, hi = lane >> 4;

  v16bf qa0 = load_a_bf16(Q + (size_t)l0 * HDIM +  0, HDIM, lane);
  v16bf qa1 = load_a_bf16(Q + (size_t)l0 * HDIM + 32, HDIM, lane);

  v8f o0 = {}, o1 = {}, o2 = {}, o3 = {};
  float mrow[8], lrow[8];
  for (int r = 0; r < 8; ++r) { mrow[r] = -3.4e38f; lrow[r] = 0.f; }

#define LOAD_KV(K00, K01, K10, K11, V0, V1, V2, V3, C0, C1, S)              \
  K00 = load_b_bf16(K + (size_t)((S) +  0) * HDIM +  0, HDIM, lane);        \
  K01 = load_b_bf16(K + (size_t)((S) +  0) * HDIM + 32, HDIM, lane);        \
  K10 = load_b_bf16(K + (size_t)((S) + 16) * HDIM +  0, HDIM, lane);        \
  K11 = load_b_bf16(K + (size_t)((S) + 16) * HDIM + 32, HDIM, lane);        \
  V0  = load_b_bf16(Vt + (size_t) 0 * SEQ + (S), SEQ, lane);                \
  V1  = load_b_bf16(Vt + (size_t)16 * SEQ + (S), SEQ, lane);                \
  V2  = load_b_bf16(Vt + (size_t)32 * SEQ + (S), SEQ, lane);                \
  V3  = load_b_bf16(Vt + (size_t)48 * SEQ + (S), SEQ, lane);                \
  for (int r = 0; r < 8; ++r) {                                             \
    size_t crow = (size_t)(l0 + r + hi * 8) * SEQ + (S);                    \
    C0[r] = comb[crow + nloc];                                              \
    C1[r] = comb[crow + 16 + nloc];                                         \
  }

  v16bf kb00, kb01, kb10, kb11, vb0, vb1, vb2, vb3;
  v8f cc0, cc1;
  LOAD_KV(kb00, kb01, kb10, kb11, vb0, vb1, vb2, vb3, cc0, cc1, 0)

#pragma unroll 2
  for (int s0 = 0; s0 < SEQ; s0 += 32) {
    // scores for current step: C initialized with mask+bias (prefetched)
    v8f sc0 = cc0, sc1 = cc1;
    sc0 = wmma_bf16(qa0, kb00, sc0);
    sc0 = wmma_bf16(qa1, kb01, sc0);
    sc1 = wmma_bf16(qa0, kb10, sc1);
    sc1 = wmma_bf16(qa1, kb11, sc1);

    // prefetch next step's K, V, comb (wraps harmlessly on last iteration)
    int sn = (s0 + 32) & (SEQ - 1);
    v16bf kn00, kn01, kn10, kn11, vn0, vn1, vn2, vn3;
    v8f cn0, cn1;
    LOAD_KV(kn00, kn01, kn10, kn11, vn0, vn1, vn2, vn3, cn0, cn1, sn)

    // online softmax: per-row stats reduced over the 16-lane column groups
    for (int r = 0; r < 8; ++r) {
      float mx = fmaxf(sc0[r], sc1[r]);
      for (int off = 1; off < 16; off <<= 1) mx = fmaxf(mx, __shfl_xor(mx, off, 32));
      float mnew  = fmaxf(mrow[r], mx);
      float alpha = __expf(mrow[r] - mnew);
      float p0 = __expf(sc0[r] - mnew);
      float p1 = __expf(sc1[r] - mnew);
      float ps = p0 + p1;
      for (int off = 1; off < 16; off <<= 1) ps += __shfl_xor(ps, off, 32);
      lrow[r] = lrow[r] * alpha + ps;
      mrow[r] = mnew;
      o0[r] *= alpha; o1[r] *= alpha; o2[r] *= alpha; o3[r] *= alpha;
      plds[wave][r + hi * 8][nloc]      = (__bf16)p0;  // C-layout -> row-major LDS
      plds[wave][r + hi * 8][16 + nloc] = (__bf16)p1;
    }
    asm volatile("s_wait_dscnt 0" ::: "memory");
    v16bf pa = load_a_bf16(&plds[wave][0][0], 40, lane);       // A-layout read-back
    o0 = wmma_bf16(pa, vb0, o0);
    o1 = wmma_bf16(pa, vb1, o1);
    o2 = wmma_bf16(pa, vb2, o2);
    o3 = wmma_bf16(pa, vb3, o3);

    kb00 = kn00; kb01 = kn01; kb10 = kn10; kb11 = kn11;
    vb0 = vn0; vb1 = vn1; vb2 = vn2; vb3 = vn3;
    cc0 = cn0; cc1 = cn1;
  }
#undef LOAD_KV

  for (int r = 0; r < 8; ++r) {
    float inv = 1.0f / lrow[r];
    int l = l0 + r + hi * 8;
    size_t base = ((size_t)b * SEQ + l) * E_DIM + h * HDIM;
    aw[base + nloc]      = (__bf16)(o0[r] * inv);
    aw[base + 16 + nloc] = (__bf16)(o1[r] * inv);
    aw[base + 32 + nloc] = (__bf16)(o2[r] * inv);
    aw[base + 48 + nloc] = (__bf16)(o3[r] * inv);
  }
}

// ---- Kernel 3: output projection (WMMA bf16, ping-pong pipelined) ------
__global__ __launch_bounds__(128) void out_proj_kernel(
    const __bf16* __restrict__ aw, const __bf16* __restrict__ Wob,
    const float* __restrict__ bo, float* __restrict__ out) {
  int lane = threadIdx.x & 31, wave = threadIdx.x >> 5;
  int m0 = (blockIdx.x * 4 + wave) * 16;
  int n0 = blockIdx.y * 64;
  const __bf16* A0 = aw  + (size_t)m0 * E_DIM;
  const __bf16* W0 = Wob + (size_t)n0 * E_DIM;

  v8f acc0 = {}, acc1 = {}, acc2 = {}, acc3 = {};

#define LOAD_SET(AV, B0, B1, B2, B3, KK)                                   \
  AV = load_a_bf16(A0 + (KK), E_DIM, lane);                                \
  B0 = load_b_bf16(W0 + (size_t) 0 * E_DIM + (KK), E_DIM, lane);           \
  B1 = load_b_bf16(W0 + (size_t)16 * E_DIM + (KK), E_DIM, lane);           \
  B2 = load_b_bf16(W0 + (size_t)32 * E_DIM + (KK), E_DIM, lane);           \
  B3 = load_b_bf16(W0 + (size_t)48 * E_DIM + (KK), E_DIM, lane);
#define DO_STEP(AV, B0, B1, B2, B3)                                        \
  acc0 = wmma_bf16(AV, B0, acc0);                                          \
  acc1 = wmma_bf16(AV, B1, acc1);                                          \
  acc2 = wmma_bf16(AV, B2, acc2);                                          \
  acc3 = wmma_bf16(AV, B3, acc3);

  v16bf aP, bP0, bP1, bP2, bP3;
  v16bf aQ, bQ0, bQ1, bQ2, bQ3;
  LOAD_SET(aP, bP0, bP1, bP2, bP3, 0)
  int kk = 32;
  for (int it = 0; it < 15; ++it, kk += 64) {
    LOAD_SET(aQ, bQ0, bQ1, bQ2, bQ3, kk)
    DO_STEP(aP, bP0, bP1, bP2, bP3)
    LOAD_SET(aP, bP0, bP1, bP2, bP3, kk + 32)
    DO_STEP(aQ, bQ0, bQ1, bQ2, bQ3)
  }
  LOAD_SET(aQ, bQ0, bQ1, bQ2, bQ3, 992)
  DO_STEP(aP, bP0, bP1, bP2, bP3)
  DO_STEP(aQ, bQ0, bQ1, bQ2, bQ3)
#undef LOAD_SET
#undef DO_STEP

  int nloc = lane & 15, hi = lane >> 4;
#define OUT_STORE(T, ACC)                                                        \
  {                                                                              \
    float bias = bo[n0 + (T) * 16 + nloc];                                       \
    for (int r = 0; r < 8; ++r)                                                  \
      out[(size_t)(m0 + r + hi * 8) * E_DIM + n0 + (T) * 16 + nloc] =            \
          ACC[r] + bias;                                                         \
  }
  OUT_STORE(0, acc0) OUT_STORE(1, acc1) OUT_STORE(2, acc2) OUT_STORE(3, acc3)
#undef OUT_STORE
}

// ---- Host-side launch ---------------------------------------------------
extern "C" void kernel_launch(void* const* d_in, const int* in_sizes, int n_in,
                              void* d_out, int out_size, void* d_ws, size_t ws_size,
                              hipStream_t stream) {
  const float* query = (const float*)d_in[0];
  const float* key   = (const float*)d_in[1];
  const float* value = (const float*)d_in[2];
  const float* abias = (const float*)d_in[3];
  const float* amask = (const float*)d_in[4];
  const float* Wqkv  = (const float*)d_in[5];
  const float* bqkv  = (const float*)d_in[6];
  const float* Wo    = (const float*)d_in[7];
  const float* bo    = (const float*)d_in[8];
  float* out = (float*)d_out;

  const size_t TILE_BYTES = (size_t)BATCH * HEADS * SEQ * HDIM * 2;  // 16 MiB bf16
  char* ws = (char*)d_ws;
  __bf16* qw      = (__bf16*)(ws + 0 * TILE_BYTES);
  __bf16* kw      = (__bf16*)(ws + 1 * TILE_BYTES);
  __bf16* vtw     = (__bf16*)(ws + 2 * TILE_BYTES);
  __bf16* attw    = (__bf16*)(ws + 3 * TILE_BYTES);
  float*  comb    = (float*) (ws + 4 * TILE_BYTES);             // 16 MiB f32
  __bf16* wqkv_bf = (__bf16*)(ws + 5 * TILE_BYTES);             // 6 MiB
  __bf16* wo_bf   = (__bf16*)(ws + 5 * TILE_BYTES + (size_t)3 * E_DIM * E_DIM * 2);

  int n4c = (SEQ * SEQ) / 4;
  combine_kernel<<<dim3((n4c + 255) / 256), 256, 0, stream>>>(abias, amask, comb, n4c);
  int n4w = (3 * E_DIM * E_DIM) / 4;
  w_to_bf16_kernel<<<dim3((n4w + 255) / 256), 256, 0, stream>>>(Wqkv, wqkv_bf, n4w);
  int n4o = (E_DIM * E_DIM) / 4;
  w_to_bf16_kernel<<<dim3((n4o + 255) / 256), 256, 0, stream>>>(Wo, wo_bf, n4o);

  qkv_proj_kernel<<<dim3(128, 16, 3), 128, 0, stream>>>(
      query, key, value, wqkv_bf, bqkv, qw, kw, vtw);
  attn_kernel<<<dim3(32, 64), 128, 0, stream>>>(qw, kw, vtw, comb, attw);
  out_proj_kernel<<<dim3(128, 16), 128, 0, stream>>>(attw, wo_bf, bo, out);
}
